// QKVAttention_1503238553906
// MI455X (gfx1250) — matrix operand order repaired and checked by reading, not compiled, not run on hardware
//
#include <hip/hip_runtime.h>
#include <math.h>

// ---------------------------------------------------------------------------
// CDNA5 (gfx1250) wave32 WMMA attention pipeline, fixed sizes:
//   B=8, N=M=2048, H=O=1024
//   lq/lk = proj -> f16 (row major) ; lvT = proj -> f16 (transposed, B x H x M)
//   S = lq@lk^T*scale (f16) ; P = masked softmax (in place) ; out = P@lv (f32)
// Attention GEMMs use double-buffered Tensor-Data-Mover loads (TENSORcnt
// pipelined against WMMA); projections use register-staged global loads.
// ---------------------------------------------------------------------------

#define HD  1024      // H == O == DQ == DK == DV
#define SEQ 2048      // N == M
#define NB  8         // batch

#define BM 64         // block tile rows
#define BN 128        // block tile cols (8 waves x 16)
#define BK 64         // K per stage (2 WMMA k-steps)
#define LS 72         // padded LDS row stride in halves (64 + 8)

typedef __attribute__((ext_vector_type(16))) _Float16 v16h;
typedef __attribute__((ext_vector_type(8)))  float    v8f;

union Frag   { v16h h;         uint4 q[2]; };
union Pack8  { _Float16 h[8];  uint4 q;    };
union Pack16 { _Float16 h[16]; uint4 q[2]; };
union Pack32 { _Float16 h[32]; uint4 q[4]; };

__device__ __forceinline__ v8f wmma_acc(const Frag& a, const Frag& b, v8f c) {
    return __builtin_amdgcn_wmma_f32_16x16x32_f16(
        false, a.h, false, b.h, (short)0, c, false, false);
}

// A fragment (16x32): lane m = lane%16, hi = lane/16
__device__ __forceinline__ void load_a_frag(Frag& f, const _Float16* As,
                                            int mrow, int ks, int hi) {
    const _Float16* p = As + mrow * LS + ks * 32 + hi * 16;
    f.q[0] = *reinterpret_cast<const uint4*>(p);
    f.q[1] = *reinterpret_cast<const uint4*>(p + 8);
}

// B fragment (32x16) from N-major LDS: lane n = lane%16, K = ks*32 + hi*16 + e
__device__ __forceinline__ void load_b_frag(Frag& f, const _Float16* Bs,
                                            int nrow, int ks, int hi) {
    const _Float16* p = Bs + nrow * LS + ks * 32 + hi * 16;
    f.q[0] = *reinterpret_cast<const uint4*>(p);
    f.q[1] = *reinterpret_cast<const uint4*>(p + 8);
}

// 8 WMMAs for one staged 64-K tile
__device__ __forceinline__ void compute_stage(const _Float16* As,
                                              const _Float16* Bs,
                                              int wave, int m16, int hi,
                                              v8f acc[4]) {
    #pragma unroll
    for (int ks = 0; ks < 2; ++ks) {
        Frag fb; load_b_frag(fb, Bs, wave * 16 + m16, ks, hi);
        #pragma unroll
        for (int mt = 0; mt < 4; ++mt) {
            Frag fa; load_a_frag(fa, As, mt * 16 + m16, ks, hi);
            acc[mt] = wmma_acc(fa, fb, acc[mt]);
        }
    }
}

// ---------------------------------------------------------------------------
// Tensor Data Mover: 2D f16 tile (lines of 64 elems) -> LDS with padding of
// 4 DWORDs every 32 DWORDs (== 72-half LDS row stride).
// ---------------------------------------------------------------------------
#if defined(__has_builtin)
#if __has_builtin(__builtin_amdgcn_tensor_load_to_lds) && \
    __has_builtin(__builtin_amdgcn_s_wait_tensorcnt)
#define HAVE_TDM 1
#endif
#endif
#ifndef HAVE_TDM
#define HAVE_TDM 0
#endif

#if HAVE_TDM
typedef unsigned int u32x4_t __attribute__((ext_vector_type(4)));
typedef int          i32x4_t __attribute__((ext_vector_type(4)));
typedef int          i32x8_t __attribute__((ext_vector_type(8)));

__device__ __forceinline__ void tdm_load_tile_f16(const _Float16* gsrc,
                                                  _Float16* ldst,
                                                  int line_stride, int tile_h) {
    unsigned long long ga = (unsigned long long)(uintptr_t)gsrc;
    unsigned int       la = (unsigned int)(uintptr_t)ldst;   // LDS byte offset

    u32x4_t g0;
    g0.x = 1u;                                            // count=1, user desc
    g0.y = la;                                            // lds_addr
    g0.z = (unsigned int)ga;                              // global_addr[31:0]
    g0.w = (unsigned int)((ga >> 32) & 0x01FFFFFFu) | 0x80000000u; // +type=2

    const unsigned int td0 = (unsigned int)line_stride;   // tensor_dim0
    const unsigned int td1 = 0x100000u;                   // tensor_dim1 (large)
    i32x8_t g1;
    g1[0] = (int)((1u << 16)      // data_size = 2 bytes
                | (1u << 20)      // pad_enable
                | (4u << 22)      // pad_interval: every 32 DWORDs (128 B line)
                | (3u << 25));    // pad_amount: 4 DWORDs (8 halves)
    g1[1] = (int)((td0 & 0xFFFFu) << 16);
    g1[2] = (int)((td0 >> 16) | ((td1 & 0xFFFFu) << 16));
    g1[3] = (int)((td1 >> 16) | (64u << 16));             // tile_dim0 = 64
    g1[4] = (int)((unsigned int)tile_h);                  // tile_dim1
    g1[5] = (int)(unsigned int)line_stride;               // dim0_stride lo32
    g1[6] = 0;
    g1[7] = 0;

    i32x4_t z4 = {0, 0, 0, 0};
#if defined(__clang_major__) && __clang_major__ >= 23
    i32x8_t z8 = {0, 0, 0, 0, 0, 0, 0, 0};
    __builtin_amdgcn_tensor_load_to_lds(g0, g1, z4, z4, z8, 0);
#else
    __builtin_amdgcn_tensor_load_to_lds(g0, g1, z4, z4, 0);
#endif
}
#endif // HAVE_TDM

// ---------------------------------------------------------------------------
// Kernel 1: projection GEMM (register-staged pipeline).
//   TRANS = 0: Y[row, col] (row-major rows x 1024)
//   TRANS = 1: YT[b, col, m] (per batch col-major, for lv^T)
// ---------------------------------------------------------------------------
template <int TRANS>
__global__ void __launch_bounds__(256)
proj_gemm(const float* __restrict__ X, const float* __restrict__ W,
          const float* __restrict__ bias, _Float16* __restrict__ Y)
{
    __shared__ __align__(32) _Float16 As[BM * LS];
    __shared__ __align__(32) _Float16 Bs[BN * LS];

    const int t    = threadIdx.x;
    const int lane = t & 31;
    const int wave = t >> 5;
    const int m16  = lane & 15;
    const int hi   = lane >> 4;
    const int row0 = blockIdx.x * BM;
    const int col0 = blockIdx.y * BN;

    v8f acc[4] = {};

    const int ar = t >> 2;           // A stage: row 0..63
    const int ac = (t & 3) * 16;     // col 0/16/32/48
    const int bn = t & 127;          // B stage: n 0..127
    const int bk = (t >> 7) * 32;    // k base 0/32

    const float* asrc = X + (size_t)(row0 + ar) * HD + ac;
    const float* bsrc = W + (size_t)bk * HD + col0 + bn;

    // ---- load tile 0 into registers, then LDS ----
    Pack16 pa; Pack32 pb;
    {
        const float4* s4 = reinterpret_cast<const float4*>(asrc);
        #pragma unroll
        for (int i = 0; i < 4; ++i) {
            float4 x = s4[i];
            pa.h[i * 4 + 0] = (_Float16)x.x; pa.h[i * 4 + 1] = (_Float16)x.y;
            pa.h[i * 4 + 2] = (_Float16)x.z; pa.h[i * 4 + 3] = (_Float16)x.w;
        }
        #pragma unroll
        for (int kk = 0; kk < 32; ++kk) pb.h[kk] = (_Float16)bsrc[kk * HD];
    }
    {
        uint4* da = reinterpret_cast<uint4*>(As + ar * LS + ac);
        da[0] = pa.q[0]; da[1] = pa.q[1];
        uint4* db = reinterpret_cast<uint4*>(Bs + bn * LS + bk);
        #pragma unroll
        for (int i = 0; i < 4; ++i) db[i] = pb.q[i];
    }

    const int NIT = HD / BK;   // 16
    for (int i = 0; i < NIT; ++i) {
        __syncthreads();                       // tile i visible in LDS
        const int k1 = (i + 1) * BK;
        Pack16 na; Pack32 nb;
        if (i + 1 < NIT) {                     // fetch tile i+1 into regs
            const float4* s4 = reinterpret_cast<const float4*>(asrc + k1);
            #pragma unroll
            for (int j = 0; j < 4; ++j) {
                float4 x = s4[j];
                na.h[j * 4 + 0] = (_Float16)x.x; na.h[j * 4 + 1] = (_Float16)x.y;
                na.h[j * 4 + 2] = (_Float16)x.z; na.h[j * 4 + 3] = (_Float16)x.w;
            }
            const float* bs = bsrc + (size_t)k1 * HD;
            #pragma unroll
            for (int kk = 0; kk < 32; ++kk) nb.h[kk] = (_Float16)bs[kk * HD];
        }
        compute_stage(As, Bs, wave, m16, hi, acc);
        __syncthreads();                       // all done reading tile i
        if (i + 1 < NIT) {
            uint4* da = reinterpret_cast<uint4*>(As + ar * LS + ac);
            da[0] = na.q[0]; da[1] = na.q[1];
            uint4* db = reinterpret_cast<uint4*>(Bs + bn * LS + bk);
            #pragma unroll
            for (int j = 0; j < 4; ++j) db[j] = nb.q[j];
        }
    }

    const int col = col0 + wave * 16 + m16;
    const float bv = bias[col];
    if (TRANS == 0) {
        #pragma unroll
        for (int mt = 0; mt < 4; ++mt)
            #pragma unroll
            for (int r = 0; r < 8; ++r) {
                int m = mt * 16 + r + hi * 8;
                Y[(size_t)(row0 + m) * HD + col] = (_Float16)(acc[mt][r] + bv);
            }
    } else {
        // transposed store: (b, col, m); block never straddles a batch
        const int bidx  = row0 / SEQ;
        const int mbase = row0 % SEQ;
        _Float16* yt = Y + ((size_t)bidx * HD + col) * SEQ + mbase + hi * 8;
        #pragma unroll
        for (int mt = 0; mt < 4; ++mt) {
            Pack8 p;
            #pragma unroll
            for (int r = 0; r < 8; ++r) p.h[r] = (_Float16)(acc[mt][r] + bv);
            *reinterpret_cast<uint4*>(yt + mt * 16) = p.q;
        }
    }
}

// ---------------------------------------------------------------------------
// Double-buffered TDM GEMM body for f16 x f16 tiles (both straight copies).
//   A: BM lines of 64, stride LDA ; B: BN lines of 64, stride LDB
// ---------------------------------------------------------------------------
__device__ __forceinline__ void tdm_gemm_body(
    const _Float16* __restrict__ A, int LDA,
    const _Float16* __restrict__ Bm, int LDB,
    int niter, int t, int wave, int m16, int hi,
    _Float16* As, _Float16* Bs, v8f acc[4])
{
#if HAVE_TDM
    if (wave == 0) {
        tdm_load_tile_f16(A,  As, LDA, BM);
        tdm_load_tile_f16(Bm, Bs, LDB, BN);
    }
    for (int i = 0; i < niter; ++i) {
        const int cur = i & 1;
        _Float16* Ac = As + cur * (BM * LS);
        _Float16* Bc = Bs + cur * (BN * LS);
        if (wave == 0) {
            if (i + 1 < niter) {
                const int k1 = (i + 1) * BK;
                tdm_load_tile_f16(A + k1,  As + (cur ^ 1) * (BM * LS), LDA, BM);
                tdm_load_tile_f16(Bm + k1, Bs + (cur ^ 1) * (BN * LS), LDB, BN);
                __builtin_amdgcn_s_wait_tensorcnt(2);  // tile i done, i+1 in flight
            } else {
                __builtin_amdgcn_s_wait_tensorcnt(0);
            }
        }
        __syncthreads();
        compute_stage(Ac, Bc, wave, m16, hi, acc);
        __syncthreads();
    }
#else
    for (int i = 0; i < niter; ++i) {
        const int cur = i & 1;
        _Float16* Ac = As + cur * (BM * LS);
        _Float16* Bc = Bs + cur * (BN * LS);
        const int k0 = i * BK;
        {
            int r = t >> 2, cb = (t & 3) * 16;
            uint4* dst = reinterpret_cast<uint4*>(Ac + r * LS + cb);
            const uint4* src = reinterpret_cast<const uint4*>(
                A + (size_t)r * LDA + k0 + cb);
            dst[0] = src[0]; dst[1] = src[1];
        }
        {
            int n = t >> 1, cb = (t & 1) * 32;
            uint4* dst = reinterpret_cast<uint4*>(Bc + n * LS + cb);
            const uint4* src = reinterpret_cast<const uint4*>(
                Bm + (size_t)n * LDB + k0 + cb);
            dst[0] = src[0]; dst[1] = src[1]; dst[2] = src[2]; dst[3] = src[3];
        }
        __syncthreads();
        compute_stage(Ac, Bc, wave, m16, hi, acc);
        __syncthreads();
    }
#endif
}

// ---------------------------------------------------------------------------
// Kernel 2: scores. S(f16) = (lq @ lk^T) * scale    (K = HD)
// ---------------------------------------------------------------------------
__global__ void __launch_bounds__(256)
attn_scores(const _Float16* __restrict__ LQ, const _Float16* __restrict__ LK,
            _Float16* __restrict__ S, float scale)
{
    __shared__ __align__(32) _Float16 As[2 * BM * LS];
    __shared__ __align__(32) _Float16 Bs[2 * BN * LS];

    const int b    = blockIdx.z;
    const int t    = threadIdx.x;
    const int lane = t & 31;
    const int wave = t >> 5;
    const int m16  = lane & 15;
    const int hi   = lane >> 4;
    const int row0 = blockIdx.x * BM;   // query rows
    const int col0 = blockIdx.y * BN;   // key rows

    v8f acc[4] = {};
    tdm_gemm_body(LQ + ((size_t)b * SEQ + row0) * HD, HD,
                  LK + ((size_t)b * SEQ + col0) * HD, HD,
                  HD / BK, t, wave, m16, hi, As, Bs, acc);

    _Float16* C = S + (size_t)b * SEQ * SEQ;
    const int col = col0 + wave * 16 + m16;
    #pragma unroll
    for (int mt = 0; mt < 4; ++mt)
        #pragma unroll
        for (int r = 0; r < 8; ++r) {
            int m = mt * 16 + r + hi * 8;
            C[(size_t)(row0 + m) * SEQ + col] = (_Float16)(acc[mt][r] * scale);
        }
}

// ---------------------------------------------------------------------------
// Kernel 3: masked softmax in place (rows of length SEQ).
// ---------------------------------------------------------------------------
__global__ void __launch_bounds__(256)
softmax_mask(_Float16* __restrict__ S, const int* __restrict__ seq_lens)
{
    const int b   = blockIdx.y;
    const int row = blockIdx.x;
    const int t   = threadIdx.x;
    _Float16* srow = S + ((size_t)b * SEQ + (size_t)row) * SEQ;
    const int len = seq_lens[b];

    float vals[8];
    float mx = -1e30f;
    #pragma unroll
    for (int i = 0; i < 8; ++i) {
        int m = t + i * 256;
        float x = (m < len) ? (float)srow[m] : -1e30f;
        vals[i] = x;
        mx = fmaxf(mx, x);
    }

    __shared__ float red[256];
    red[t] = mx; __syncthreads();
    for (int s = 128; s > 0; s >>= 1) {
        if (t < s) red[t] = fmaxf(red[t], red[t + s]);
        __syncthreads();
    }
    const float gmax = red[0];
    __syncthreads();

    float sum = 0.f;
    #pragma unroll
    for (int i = 0; i < 8; ++i) {
        int m = t + i * 256;
        float e = (m < len) ? __expf(vals[i] - gmax) : 0.f;
        vals[i] = e;
        sum += e;
    }
    red[t] = sum; __syncthreads();
    for (int s = 128; s > 0; s >>= 1) {
        if (t < s) red[t] += red[t + s];
        __syncthreads();
    }
    const float inv = 1.f / red[0];

    #pragma unroll
    for (int i = 0; i < 8; ++i) {
        int m = t + i * 256;
        srow[m] = (_Float16)(vals[i] * inv);
    }
}

// ---------------------------------------------------------------------------
// Kernel 4: out(f32) = P(f16, SEQxSEQ) @ lv ; lv given transposed (B, HD, SEQ)
// Both tiles straight copies -> fully TDM double-buffered.
// ---------------------------------------------------------------------------
__global__ void __launch_bounds__(256)
attn_out(const _Float16* __restrict__ P, const _Float16* __restrict__ LVT,
         float* __restrict__ Out)
{
    __shared__ __align__(32) _Float16 As[2 * BM * LS];
    __shared__ __align__(32) _Float16 Bs[2 * BN * LS];

    const int b    = blockIdx.z;
    const int t    = threadIdx.x;
    const int lane = t & 31;
    const int wave = t >> 5;
    const int m16  = lane & 15;
    const int hi   = lane >> 4;
    const int row0 = blockIdx.x * BM;
    const int col0 = blockIdx.y * BN;   // output feature block

    v8f acc[4] = {};
    tdm_gemm_body(P   + ((size_t)b * SEQ + row0) * SEQ, SEQ,
                  LVT + ((size_t)b * HD  + col0) * SEQ, SEQ,
                  SEQ / BK, t, wave, m16, hi, As, Bs, acc);

    float* C = Out + (size_t)b * SEQ * HD;
    const int col = col0 + wave * 16 + m16;
    #pragma unroll
    for (int mt = 0; mt < 4; ++mt)
        #pragma unroll
        for (int r = 0; r < 8; ++r) {
            int m = mt * 16 + r + hi * 8;
            C[(size_t)(row0 + m) * HD + col] = acc[mt][r];
        }
}

// ---------------------------------------------------------------------------
extern "C" void kernel_launch(void* const* d_in, const int* in_sizes, int n_in,
                              void* d_out, int out_size, void* d_ws, size_t ws_size,
                              hipStream_t stream) {
    (void)in_sizes; (void)n_in; (void)out_size; (void)ws_size;

    const float* q   = (const float*)d_in[0];
    const float* k   = (const float*)d_in[1];
    const float* v   = (const float*)d_in[2];
    const int*   seq = (const int*)  d_in[3];
    const float* Wq  = (const float*)d_in[4];
    const float* bq  = (const float*)d_in[5];
    const float* Wk  = (const float*)d_in[6];
    const float* bk  = (const float*)d_in[7];
    const float* Wv  = (const float*)d_in[8];
    const float* bv  = (const float*)d_in[9];
    float* out = (float*)d_out;

    // workspace (f16): lq | lk | lvT | S/P
    _Float16* lq  = (_Float16*)d_ws;
    _Float16* lk  = lq  + (size_t)NB * SEQ * HD;
    _Float16* lvT = lk  + (size_t)NB * SEQ * HD;
    _Float16* S   = lvT + (size_t)NB * SEQ * HD;

    dim3 blk(256);

    dim3 gp((NB * SEQ) / BM, HD / BN);
    proj_gemm<0><<<gp, blk, 0, stream>>>(q, Wq, bq, lq);
    proj_gemm<0><<<gp, blk, 0, stream>>>(k, Wk, bk, lk);
    proj_gemm<1><<<gp, blk, 0, stream>>>(v, Wv, bv, lvT);

    dim3 gs(SEQ / BM, SEQ / BN, NB);
    attn_scores<<<gs, blk, 0, stream>>>(lq, lk, S, 0.03125f);

    dim3 gm(SEQ, NB);
    softmax_mask<<<gm, blk, 0, stream>>>(S, seq);

    dim3 go(SEQ / BM, HD / BN, NB);
    attn_out<<<go, blk, 0, stream>>>(S, lvT, out);
}
